// TMLMamba_50259707297890
// MI455X (gfx1250) — matrix operand order repaired and checked by reading, not compiled, not run on hardware
//
#include <hip/hip_runtime.h>
#include <hip/hip_bf16.h>

typedef __attribute__((ext_vector_type(16))) _Float16 v16h;
typedef __attribute__((ext_vector_type(8)))  _Float16 v8h;
typedef __attribute__((ext_vector_type(2)))  __fp16   h2raw;   // return type of cvt_pkrtz
typedef __attribute__((ext_vector_type(8)))  float    v8f;
typedef __attribute__((ext_vector_type(4)))  float    v4f;

#define CB 16
#define CS 768
#define CD 512
#define CN 64
#define CL 6
#define CFF 2048

#define FLAG_RELU 1
#define FLAG_ACC  2
#define FLAG_RES  4
#define FLAG_BIAS 8

#define BM 128
#define BN 64
#define KT 32
#define ASTR 40   // padded LDS row stride in halves (16B aligned rows)

// ---------------------------------------------------------------------------
// Generic WMMA GEMM: C[M,N] (+)= act( A_shifted[M,K] * B[K,N] + bias ) + res
//   A rows are (b,s) with s' = s + shift (zero outside [0,seg))  -> dilated conv taps
//   B element (k,n) at Bw[bbase + k*bsk + n*bsn]                 -> any weight layout
// 256 threads = 8 waves; block tile 128x64; wave tile 32x32 (4 WMMA / K-step).
// Software-pipelined: global loads for step k+KT issued while step k computes.
// ---------------------------------------------------------------------------
__global__ __launch_bounds__(256) void wmma_gemm_kernel(
    const float* __restrict__ A, int lda, int shift, int seg,
    const float* __restrict__ Bw, long bbase, int bsk, int bsn,
    const float* __restrict__ bias, const float* __restrict__ res,
    int res_mod, int res_ld,
    float* __restrict__ C, int ldc, int K, int flags)
{
  __shared__ __attribute__((aligned(16))) _Float16 As[BM * ASTR];
  __shared__ __attribute__((aligned(16))) _Float16 Bs[BN * ASTR];

  const int tid  = threadIdx.x;
  const int wave = tid >> 5;
  const int lane = tid & 31;
  const int wm   = wave & 3;          // 0..3 : 32-row slice
  const int wn   = wave >> 2;         // 0..1 : 32-col slice
  const int m0   = blockIdx.y * BM;
  const int n0   = blockIdx.x * BN;

  v8f acc00 = {}, acc01 = {}, acc10 = {}, acc11 = {};

  // A staging: thread -> (row = tid/2, 16 contiguous cols at (tid&1)*16)
  const int arow = tid >> 1;
  const int acol = (tid & 1) * 16;
  const int gm = m0 + arow;
  const int b  = gm / seg;
  const int s  = gm % seg;
  const int s2 = s + shift;
  const bool avalid = (s2 >= 0) && (s2 < seg);
  const float* Aptr = A + (long)(b * seg + s2) * lda + acol;

  // B staging: thread -> (k = tid/8, 8 contiguous n at (tid&7)*8); stored Bs[n][k]
  const int bk = tid >> 3;
  const int bn = (tid & 7) * 8;
  const float* Bptr = Bw + bbase + (long)bk * bsk + (long)(n0 + bn) * bsn;

  v4f  areg0 = {}, areg1 = {}, areg2 = {}, areg3 = {};
  float breg[8];

  // ---- prologue: stage k0 = 0 ----
  if (avalid) {
    const v4f* p = (const v4f*)Aptr;
    areg0 = p[0]; areg1 = p[1]; areg2 = p[2]; areg3 = p[3];
  }
  if (bsn == 1) {
    const v4f* q = (const v4f*)Bptr;
    v4f q0 = q[0], q1 = q[1];
#pragma unroll
    for (int j = 0; j < 4; ++j) { breg[j] = q0[j]; breg[4 + j] = q1[j]; }
  } else {
#pragma unroll
    for (int j = 0; j < 8; ++j) breg[j] = Bptr[(long)j * bsn];
  }

  for (int k0 = 0; k0 < K; k0 += KT) {
    // ---- commit staged tile to LDS (packed f16, 16B stores for A) ----
    union { v8h v; h2raw p[4]; } u0, u1;
    u0.p[0] = __builtin_amdgcn_cvt_pkrtz(areg0[0], areg0[1]);
    u0.p[1] = __builtin_amdgcn_cvt_pkrtz(areg0[2], areg0[3]);
    u0.p[2] = __builtin_amdgcn_cvt_pkrtz(areg1[0], areg1[1]);
    u0.p[3] = __builtin_amdgcn_cvt_pkrtz(areg1[2], areg1[3]);
    u1.p[0] = __builtin_amdgcn_cvt_pkrtz(areg2[0], areg2[1]);
    u1.p[1] = __builtin_amdgcn_cvt_pkrtz(areg2[2], areg2[3]);
    u1.p[2] = __builtin_amdgcn_cvt_pkrtz(areg3[0], areg3[1]);
    u1.p[3] = __builtin_amdgcn_cvt_pkrtz(areg3[2], areg3[3]);
    *(v8h*)(&As[arow * ASTR + acol])     = u0.v;
    *(v8h*)(&As[arow * ASTR + acol + 8]) = u1.v;
#pragma unroll
    for (int j = 0; j < 8; ++j)
      Bs[(bn + j) * ASTR + bk] = (_Float16)breg[j];

    __syncthreads();

    // ---- issue global loads for next K-step (overlaps with WMMA below) ----
    const int kn = k0 + KT;
    if (kn < K) {
      if (avalid) {
        const v4f* p = (const v4f*)(Aptr + kn);
        areg0 = p[0]; areg1 = p[1]; areg2 = p[2]; areg3 = p[3];
      }
      if (bsn == 1) {
        const v4f* q = (const v4f*)(Bptr + (long)kn * bsk);
        v4f q0 = q[0], q1 = q[1];
#pragma unroll
        for (int j = 0; j < 4; ++j) { breg[j] = q0[j]; breg[4 + j] = q1[j]; }
      } else {
        const float* q = Bptr + (long)kn * bsk;
#pragma unroll
        for (int j = 0; j < 8; ++j) breg[j] = q[(long)j * bsn];
      }
    }

    // ---- fragments (ISA 7.12.2 layouts) + 4 WMMAs ----
    const int hi = lane >> 4;
    const int mr = wm * 32 + (lane & 15);
    union { v16h v; unsigned u[8]; } af0, af1, bf0, bf1;
    const unsigned* ar0 = (const unsigned*)&As[mr * ASTR];
    const unsigned* ar1 = (const unsigned*)&As[(mr + 16) * ASTR];
#pragma unroll
    for (int r = 0; r < 8; ++r) {
      const int kk = 2 * (r & 3) + 8 * hi + 16 * (r >> 2);
      af0.u[r] = ar0[kk >> 1];
      af1.u[r] = ar1[kk >> 1];
    }
    const int nr = wn * 32 + (lane & 15);
    const unsigned* br0 = (const unsigned*)&Bs[nr * ASTR];
    const unsigned* br1 = (const unsigned*)&Bs[(nr + 16) * ASTR];
#pragma unroll
    for (int r = 0; r < 8; ++r) {
      const int kk = 2 * (r & 3) + 8 * hi + 16 * (r >> 2);
      bf0.u[r] = br0[kk >> 1];
      bf1.u[r] = br1[kk >> 1];
    }

    acc00 = __builtin_amdgcn_wmma_f32_16x16x32_f16(false, af0.v, false, bf0.v,
                                                   (short)0, acc00, false, false);
    acc01 = __builtin_amdgcn_wmma_f32_16x16x32_f16(false, af0.v, false, bf1.v,
                                                   (short)0, acc01, false, false);
    acc10 = __builtin_amdgcn_wmma_f32_16x16x32_f16(false, af1.v, false, bf0.v,
                                                   (short)0, acc10, false, false);
    acc11 = __builtin_amdgcn_wmma_f32_16x16x32_f16(false, af1.v, false, bf1.v,
                                                   (short)0, acc11, false, false);

    __syncthreads();
  }

  // ---- epilogue ----
  const int ehi   = lane >> 4;
  const int mbase = m0 + wm * 32 + 8 * ehi;
  const int nA    = n0 + wn * 32 + (lane & 15);

  auto epi = [&](v8f acc, int mb, int nn) {
#pragma unroll
    for (int v = 0; v < 8; ++v) {
      const int m = mb + v;
      float val = acc[v];
      const long idx = (long)m * ldc + nn;
      if (flags & FLAG_BIAS) val += bias[nn];
      if (flags & FLAG_ACC)  val += C[idx];
      if (flags & FLAG_RELU) val = fmaxf(val, 0.0f);
      if (flags & FLAG_RES)  val += res[(long)(m % res_mod) * res_ld + nn];
      C[idx] = val;
    }
  };
  epi(acc00, mbase,      nA);
  epi(acc01, mbase,      nA + 16);
  epi(acc10, mbase + 16, nA);
  epi(acc11, mbase + 16, nA + 16);
}

// ---------------------------------------------------------------------------
// Elementwise / reduction kernels
// ---------------------------------------------------------------------------

__global__ void norm_rows_kernel(const float* __restrict__ x, float* __restrict__ z)
{
  const long m = blockIdx.x;
  const int  t = threadIdx.x;                 // 64 threads
  const float v = x[m * 64 + t];
  __shared__ float red[64];
  red[t] = v * v;
  __syncthreads();
  for (int off = 32; off > 0; off >>= 1) {
    if (t < off) red[t] += red[t + off];
    __syncthreads();
  }
  const float nrm = fmaxf(sqrtf(red[0]), 1e-12f);
  z[m * 64 + t] = v / nrm;
}

__global__ void env_mean_kernel(const int* __restrict__ idx,
                                const float* __restrict__ emb,
                                float* __restrict__ out)
{
  const long m = blockIdx.x;
  __shared__ int id[8];
  if (threadIdx.x < 8) id[threadIdx.x] = idx[m * 8 + threadIdx.x];
  __syncthreads();
  for (int d = threadIdx.x; d < CD; d += blockDim.x) {
    float s = 0.0f;
#pragma unroll
    for (int j = 0; j < 8; ++j) s += emb[(long)id[j] * CD + d];
    out[m * CD + d] = s * 0.125f;
  }
}

__global__ void mean_s_kernel(const float* __restrict__ h, float* __restrict__ out)
{
  const int b = blockIdx.y;
  const int d = blockIdx.x * blockDim.x + threadIdx.x;
  float s = 0.0f;
  for (int t = 0; t < CS; ++t) s += h[((long)b * CS + t) * CD + d];
  out[b * CD + d] = s * (1.0f / CS);
}

__global__ void alpha_kernel(const float* __restrict__ xm, const float* __restrict__ w,
                             const float* __restrict__ bias, float* __restrict__ alpha)
{
  const int b = blockIdx.x;
  const int t = threadIdx.x;                  // 128 threads
  float l[4] = {0.f, 0.f, 0.f, 0.f};
  for (int d = t; d < CD; d += 128) {
    const float x = xm[b * CD + d];
#pragma unroll
    for (int i = 0; i < 4; ++i) l[i] += x * w[d * 4 + i];
  }
  __shared__ float red[4][128];
#pragma unroll
  for (int i = 0; i < 4; ++i) red[i][t] = l[i];
  __syncthreads();
  for (int off = 64; off > 0; off >>= 1) {
    if (t < off)
      for (int i = 0; i < 4; ++i) red[i][t] += red[i][t + off];
    __syncthreads();
  }
  if (t == 0) {
    float lg[4], mx = -1e30f;
    for (int i = 0; i < 4; ++i) { lg[i] = red[i][0] + bias[i]; mx = fmaxf(mx, lg[i]); }
    float den = 0.f;
    for (int i = 0; i < 4; ++i) { lg[i] = expf(lg[i] - mx); den += lg[i]; }
    for (int i = 0; i < 4; ++i) alpha[b * 4 + i] = lg[i] / den;
  }
}

__global__ void combine_ln_kernel(const float* __restrict__ conv,
                                  const float* __restrict__ h,
                                  const float* __restrict__ alpha,
                                  const float* __restrict__ g,
                                  const float* __restrict__ be,
                                  float* __restrict__ out, long M)
{
  const long m = blockIdx.x;
  const int  t = threadIdx.x;                 // 256 threads, 2 elems each
  const int  b = (int)(m / CS);
  const float a0 = alpha[b * 4 + 0], a1 = alpha[b * 4 + 1];
  const float a2 = alpha[b * 4 + 2], a3 = alpha[b * 4 + 3];
  const long MD = M * CD;
  float v[2];
#pragma unroll
  for (int j = 0; j < 2; ++j) {
    const int d = t + j * 256;
    const long o = m * CD + d;
    v[j] = a0 * conv[o] + a1 * conv[MD + o] + a2 * conv[2 * MD + o]
         + a3 * conv[3 * MD + o] + h[o];
  }
  __shared__ float rs[256], rq[256];
  rs[t] = v[0] + v[1];
  rq[t] = v[0] * v[0] + v[1] * v[1];
  __syncthreads();
  for (int off = 128; off > 0; off >>= 1) {
    if (t < off) { rs[t] += rs[t + off]; rq[t] += rq[t + off]; }
    __syncthreads();
  }
  const float mean = rs[0] * (1.0f / CD);
  const float var  = rq[0] * (1.0f / CD) - mean * mean;
  const float inv  = rsqrtf(var + 1e-5f);
#pragma unroll
  for (int j = 0; j < 2; ++j) {
    const int d = t + j * 256;
    out[m * CD + d] = (v[j] - mean) * inv * g[d] + be[d];
  }
}

__global__ void ln_y_kernel(const float* __restrict__ h1, const float* __restrict__ y,
                            const float* __restrict__ g, const float* __restrict__ be,
                            float* __restrict__ out)
{
  const long m = blockIdx.x;
  const int  t = threadIdx.x;
  const float yv = y[m];
  float v[2];
#pragma unroll
  for (int j = 0; j < 2; ++j) v[j] = h1[m * CD + t + j * 256] + yv;
  __shared__ float rs[256], rq[256];
  rs[t] = v[0] + v[1];
  rq[t] = v[0] * v[0] + v[1] * v[1];
  __syncthreads();
  for (int off = 128; off > 0; off >>= 1) {
    if (t < off) { rs[t] += rs[t + off]; rq[t] += rq[t + off]; }
    __syncthreads();
  }
  const float mean = rs[0] * (1.0f / CD);
  const float var  = rq[0] * (1.0f / CD) - mean * mean;
  const float inv  = rsqrtf(var + 1e-5f);
#pragma unroll
  for (int j = 0; j < 2; ++j) {
    const int d = t + j * 256;
    out[m * CD + d] = (v[j] - mean) * inv * g[d] + be[d];
  }
}

__global__ void ssm_scan_kernel(const float* __restrict__ proj,
                                const float* __restrict__ h1,
                                const float* __restrict__ alog,
                                float* __restrict__ y, long M)
{
  const int b = blockIdx.x;
  const int n = threadIdx.x;                  // 64 threads (2 waves)
  const float A = alog[n];
  const long MN = M * CN;
  float hs = 0.0f;
  __shared__ float part[2];
  for (int s = 0; s < CS; ++s) {
    const long m = (long)b * CS + s;
    const float dp = proj[m * CN + n];
    const float delta = (dp > 20.0f) ? dp : log1pf(expf(dp));
    const float a  = expf(A * delta);
    const float bu = proj[MN + m * CN + n] * h1[m * CD + n];
    hs = a * hs + bu;
    float c = proj[2 * MN + m * CN + n] * hs;
    for (int off = 16; off > 0; off >>= 1) c += __shfl_down(c, off, 32);
    if ((n & 31) == 0) part[n >> 5] = c;
    __syncthreads();
    if (n == 0) y[m] = part[0] + part[1];
    __syncthreads();
  }
}

__global__ void head_kernel(const float* __restrict__ hg,
                            const float* __restrict__ w1, const float* __restrict__ b1,
                            const float* __restrict__ w2, const float* __restrict__ b2,
                            float* __restrict__ out)
{
  const int b = blockIdx.x;
  const int t = threadIdx.x;                  // 256 threads, one hidden unit each
  float acc = 0.0f;
  for (int d = 0; d < CD; ++d) acc += hg[b * CD + d] * w1[d * 256 + t];
  acc = fmaxf(acc + b1[t], 0.0f);
  __shared__ float red[256];
  red[t] = acc * w2[t];
  __syncthreads();
  for (int off = 128; off > 0; off >>= 1) {
    if (t < off) red[t] += red[t + off];
    __syncthreads();
  }
  if (t == 0) out[b] = red[0] + b2[0];
}

// ---------------------------------------------------------------------------
// Host orchestration
// ---------------------------------------------------------------------------
static inline void launch_gemm(hipStream_t st,
                               const float* A, int lda, int shift, int seg,
                               const float* Bw, long bbase, int bsk, int bsn,
                               const float* bias, const float* res, int res_mod, int res_ld,
                               float* C, int ldc, long M_, int N_, int K_, int flags)
{
  dim3 grid((unsigned)(N_ / BN), (unsigned)(M_ / BM));
  wmma_gemm_kernel<<<grid, 256, 0, st>>>(A, lda, shift, seg, Bw, bbase, bsk, bsn,
                                         bias, res, res_mod, res_ld, C, ldc, K_, flags);
}

extern "C" void kernel_launch(void* const* d_in, const int* in_sizes, int n_in,
                              void* d_out, int out_size, void* d_ws, size_t ws_size,
                              hipStream_t stream)
{
  (void)in_sizes; (void)n_in; (void)out_size; (void)ws_size;
  const float* x_cog   = (const float*)d_in[0];
  const float* cw1     = (const float*)d_in[1];
  const float* cb1     = (const float*)d_in[2];
  const float* cw2     = (const float*)d_in[3];
  const float* cb2     = (const float*)d_in[4];
  const float* env_emb = (const float*)d_in[5];
  const float* ew1     = (const float*)d_in[6];
  const float* eb1     = (const float*)d_in[7];
  const float* ew2     = (const float*)d_in[8];
  const float* eb2     = (const float*)d_in[9];
  const float* fw1     = (const float*)d_in[10];
  const float* fb1     = (const float*)d_in[11];
  const float* fw2     = (const float*)d_in[12];
  const float* fb2     = (const float*)d_in[13];
  const float* pos_enc = (const float*)d_in[14];
  const float* conv_w  = (const float*)d_in[15];
  const float* conv_b  = (const float*)d_in[16];
  const float* wm_w    = (const float*)d_in[17];
  const float* wm_b    = (const float*)d_in[18];
  const float* ssm_w   = (const float*)d_in[19];
  const float* ssm_b   = (const float*)d_in[20];
  const float* A_log   = (const float*)d_in[21];
  const float* ffw1    = (const float*)d_in[22];
  const float* ffb1    = (const float*)d_in[23];
  const float* ffw2    = (const float*)d_in[24];
  const float* ffb2    = (const float*)d_in[25];
  const float* g1      = (const float*)d_in[26];
  const float* be1     = (const float*)d_in[27];
  const float* g2      = (const float*)d_in[28];
  const float* be2     = (const float*)d_in[29];
  const float* pw1     = (const float*)d_in[30];
  const float* pb1     = (const float*)d_in[31];
  const float* pw2     = (const float*)d_in[32];
  const float* pb2     = (const float*)d_in[33];
  const int*   x_env   = (const int*)d_in[34];
  float* outp = (float*)d_out;

  const long M = (long)CB * CS;               // 12288

  float* ws = (float*)d_ws;
  float* T  = ws;                  // M*2048 : conv outs (4 x M*512) / FF hidden / temps
  float* H  = T + M * 2048;        // M*512  : residual stream
  float* Z  = H + M * CD;          // M*512  : z path, then h1
  float* E  = Z + M * CD;          // M*512  : e path, then h2
  float* P  = E + M * CD;          // 3*M*64 : proj (front-end reuses first M*64 for z64)
  float* Y  = P + 3 * M * CN;      // M      : scan output
  float* SM = Y + M;               // B*512 mean | B*4 alpha | B*512 pooled
  float* AL = SM + CB * CD;
  float* HG = AL + CB * 4;

  float* T2 = T + M * CD;          // second M*512 temp inside T

  // ---------------- Front end ----------------
  norm_rows_kernel<<<(unsigned)M, 64, 0, stream>>>(x_cog, P);
  launch_gemm(stream, P, 64, 0, CS, cw1, 0, CD, 1, cb1, nullptr, 1, 1,
              T, CD, M, CD, 64, FLAG_BIAS | FLAG_RELU);
  launch_gemm(stream, T, CD, 0, CS, cw2, 0, CD, 1, cb2, nullptr, 1, 1,
              Z, CD, M, CD, CD, FLAG_BIAS);

  env_mean_kernel<<<(unsigned)M, 256, 0, stream>>>(x_env, env_emb, T);
  launch_gemm(stream, T, CD, 0, CS, ew1, 0, CD, 1, eb1, nullptr, 1, 1,
              T2, CD, M, CD, CD, FLAG_BIAS | FLAG_RELU);
  launch_gemm(stream, T2, CD, 0, CS, ew2, 0, CD, 1, eb2, nullptr, 1, 1,
              E, CD, M, CD, CD, FLAG_BIAS);

  launch_gemm(stream, Z, CD, 0, CS, fw1, 0, CD, 1, fb1, nullptr, 1, 1,
              T, CD, M, CD, CD, FLAG_BIAS);
  launch_gemm(stream, E, CD, 0, CS, fw1, (long)CD * CD, CD, 1, nullptr, nullptr, 1, 1,
              T, CD, M, CD, CD, FLAG_ACC | FLAG_RELU);
  launch_gemm(stream, T, CD, 0, CS, fw2, 0, CD, 1, fb2, pos_enc, CS, CD,
              H, CD, M, CD, CD, FLAG_BIAS | FLAG_RES);

  // ---------------- Layers ----------------
  const int dil[4] = {1, 2, 4, 8};
  for (int l = 0; l < CL; ++l) {
    for (int i = 0; i < 4; ++i) {
      float* Ci = T + (long)i * M * CD;
      const float* bi = conv_b + ((long)l * 4 + i) * CD;
      for (int k = 0; k < 3; ++k) {
        const long bbase = ((long)l * 4 + i) * (long)CD * CD * 3 + k;
        launch_gemm(stream, H, CD, (k - 1) * dil[i], CS,
                    conv_w, bbase, 3, CD * 3,
                    bi, nullptr, 1, 1, Ci, CD, M, CD, CD,
                    (k == 0) ? FLAG_BIAS : FLAG_ACC);
      }
    }
    mean_s_kernel<<<dim3(2, CB), 256, 0, stream>>>(H, SM);
    alpha_kernel<<<CB, 128, 0, stream>>>(SM, wm_w + (long)l * CD * 4, wm_b + l * 4, AL);
    combine_ln_kernel<<<(unsigned)M, 256, 0, stream>>>(T, H, AL,
        g1 + (long)l * CD, be1 + (long)l * CD, Z, M);
    for (int k = 0; k < 3; ++k) {
      launch_gemm(stream, Z, CD, 0, CS,
                  ssm_w, ((long)l * 3 + k) * CD * CN, CN, 1,
                  ssm_b + ((long)l * 3 + k) * CN, nullptr, 1, 1,
                  P + (long)k * M * CN, CN, M, CN, CD, FLAG_BIAS);
    }
    ssm_scan_kernel<<<CB, 64, 0, stream>>>(P, Z, A_log + (long)l * CN, Y, M);
    ln_y_kernel<<<(unsigned)M, 256, 0, stream>>>(Z, Y,
        g2 + (long)l * CD, be2 + (long)l * CD, E);
    launch_gemm(stream, E, CD, 0, CS, ffw1, (long)l * CD * CFF, CFF, 1,
                ffb1 + (long)l * CFF, nullptr, 1, 1,
                T, CFF, M, CFF, CD, FLAG_BIAS | FLAG_RELU);
    launch_gemm(stream, T, CFF, 0, CS, ffw2, (long)l * CFF * CD, CD, 1,
                ffb2 + (long)l * CD, E, (int)M, CD,
                H, CD, M, CD, CFF, FLAG_BIAS | FLAG_RES);
  }

  // ---------------- Head ----------------
  mean_s_kernel<<<dim3(2, CB), 256, 0, stream>>>(H, HG);
  head_kernel<<<CB, 256, 0, stream>>>(HG, pw1, pb1, pw2, pb2, outp);
}
